// recurrent_noisy_66958540145073
// MI455X (gfx1250) — matrix operand order repaired and checked by reading, not compiled, not run on hardware
//
#include <hip/hip_runtime.h>
#include <hip/hip_bf16.h>

// ---------------------------------------------------------------------------
// recurrent noisy RNN for MI455X (gfx1250, wave32, WMMA)
//   B=256, T=512, H=512, O=2
//   scan:  h = relu(x_proj + h @ W_hh^T + b_hh + sigma*noise)
//   core:  v_wmma_f32_16x16x32_bf16; h double-buffered in LDS (2x16 KB);
//          B (packed W_hh^T, L2-resident) streamed through a 2-stage VGPR
//          ping-pong pipeline. The k-loop start is rotated by t so the B
//          addresses are t-dependent: this blocks LICM from hoisting the
//          whole W slice into registers (which spills to scratch, as seen
//          in rounds 2-3). One workgroup barrier per timestep.
// ---------------------------------------------------------------------------

typedef __bf16 v16bf  __attribute__((ext_vector_type(16)));
typedef __bf16 bf16x8 __attribute__((ext_vector_type(8)));
typedef float  v8f    __attribute__((ext_vector_type(8)));

#define B_ 256
#define T_ 512
#define H_ 512
#define O_ 2

// flat offsets (in floats) of the tuple outputs inside d_out
#define OFF_OUT 0
#define OFF_HID (B_ * O_)                        // 512
#define OFF_NET (OFF_HID + B_ * H_)              // 131584
#define OFF_RO  (OFF_NET + (size_t)B_ * T_ * H_) // 67240448

// ---------------------------------------------------------------------------
// Pack W_hh (f32 row-major [H,H]; out[m,n] = sum_k h[m,k]*W_hh[n,k]) into a
// fragment-major bf16 array so each lane's 16 B-matrix values are contiguous.
// B k order used here (contiguous, A loads match it): lane L -> column n = L%16,
// K = kt*32 + (L<16 ? 0 : 16) + e, e = 0..15 packed 2-per-VGPR.
// Wp index = (((nt*16 + kt)*32 + L)*16 + e), nt in [0,32), kt in [0,16).
// ---------------------------------------------------------------------------
__global__ void pack_whh_kernel(const float* __restrict__ Whh,
                                __bf16* __restrict__ Wp) {
    int idx = blockIdx.x * blockDim.x + threadIdx.x;   // 262144 total
    int e  = idx & 15;
    int L  = (idx >> 4) & 31;
    int kt = (idx >> 9) & 15;
    int nt = idx >> 13;
    int n = nt * 16 + (L & 15);
    int K = kt * 32 + ((L < 16) ? 0 : 16) + e;
    Wp[idx] = (__bf16)Whh[n * H_ + K];
}

// ---------------------------------------------------------------------------
// Recurrent scan. One block = 16 batch rows, 8 waves; wave w owns hidden
// columns [w*64, w*64+64). The whole T loop runs inside the block.
// ---------------------------------------------------------------------------
__launch_bounds__(256, 1)
__global__ void rnn_scan_kernel(const float* __restrict__ input,  // [B,T,2]
                                const float* __restrict__ sigma,  // [1]
                                const float* __restrict__ noise,  // [B,T,H]
                                const float* __restrict__ Win,    // [H,2]
                                const float* __restrict__ bin,    // [H]
                                const float* __restrict__ bhh,    // [H]
                                const __bf16* __restrict__ Wp,    // packed W_hh
                                float* __restrict__ net)          // [B,T,H]
{
    __shared__ __attribute__((aligned(16))) __bf16 hbuf[2][16 * H_]; // 32 KB
    __shared__ float xin_s[2][32];    // input[b0+m][t][0..1], staged 1 ahead
    __shared__ float win_s[H_ * 2];
    __shared__ float bin_s[H_];
    __shared__ float bhh_s[H_];

    const int tid  = threadIdx.x;
    const int lane = tid & 31;
    const int wv   = tid >> 5;            // 0..7
    const int b0   = blockIdx.x * 16;
    const float sig = sigma[0];

    for (int i = tid; i < H_; i += 256) {
        win_s[i * 2 + 0] = Win[i * 2 + 0];
        win_s[i * 2 + 1] = Win[i * 2 + 1];
        bin_s[i] = bin[i];
        bhh_s[i] = bhh[i];
    }
    for (int i = tid; i < 16 * H_; i += 256) hbuf[0][i] = (__bf16)0.0f;
    if (tid < 32) {  // prime step-0 inputs
        xin_s[0][tid] =
            input[(size_t)(b0 + (tid >> 1)) * (T_ * 2) + (tid & 1)];
    }
    __syncthreads();

    // Per-j base pointers into the packed B stream (this wave's 4 n-tiles).
    const __bf16* bbase[4];
    #pragma unroll
    for (int j = 0; j < 4; ++j)
        bbase[j] = Wp + (((size_t)(wv * 4 + j) * 16) * 32 + lane) * 16;

    // A-fragment addressing (16-bit A 16x32): m = lane%16,
// k = kt*32 + 0..15 (lanes 0-15) or kt*32 + 16..31 (lanes 16-31)
    const int mrow   = lane & 15;
    const int k_half = (lane < 16) ? 0 : 16;
    const int mb     = (lane < 16) ? 0 : 8;  // C/D: m = r + mb, n = lane%16

    for (int t = 0; t < T_; ++t) {
        const __bf16* hc = hbuf[t & 1];         // read h(t)
        __bf16*       hn = hbuf[(t + 1) & 1];   // write h(t+1)
        const int rot = t & 15;  // k-slab rotation: defeats LICM on B loads

        if (tid < 32 && t + 1 < T_) {  // stage next step's 2-wide input rows
            xin_s[(t + 1) & 1][tid] =
                input[(size_t)(b0 + (tid >> 1)) * (T_ * 2)
                      + (t + 1) * 2 + (tid & 1)];
        }

        // Hoist this step's noise values; latency overlaps the whole GEMM.
        float nz[4][8];
        #pragma unroll
        for (int j = 0; j < 4; ++j) {
            const int n = wv * 64 + j * 16 + (lane & 15);
            #pragma unroll
            for (int r = 0; r < 8; ++r) {
                const size_t gi =
                    ((size_t)(b0 + r + mb) * T_ + t) * H_ + n;
                nz[j][r] = noise[gi];
            }
        }

        // 2-stage ping-pong pipeline on the B stream (rotated k order).
        v16bf bbuf2[2][4];
        #pragma unroll
        for (int j = 0; j < 4; ++j) {
            const __bf16* bp = bbase[j] + (size_t)rot * 512;
            ((bf16x8*)&bbuf2[0][j])[0] = ((const bf16x8*)bp)[0];
            ((bf16x8*)&bbuf2[0][j])[1] = ((const bf16x8*)bp)[1];
        }

        v8f acc[4] = {v8f{}, v8f{}, v8f{}, v8f{}};

        #pragma unroll
        for (int kt = 0; kt < 16; ++kt) {
            const int cur = kt & 1, nxt = cur ^ 1;
            if (kt + 1 < 16) {  // prefetch next (rotated) k-slab of B
                const int ke_n = (kt + 1 + rot) & 15;
                #pragma unroll
                for (int j = 0; j < 4; ++j) {
                    const __bf16* bp = bbase[j] + (size_t)ke_n * 512;
                    ((bf16x8*)&bbuf2[nxt][j])[0] = ((const bf16x8*)bp)[0];
                    ((bf16x8*)&bbuf2[nxt][j])[1] = ((const bf16x8*)bp)[1];
                }
            }
            // A: h(t) tile from LDS (two ds_load_b128 per lane), same k order
            const int ke = (kt + rot) & 15;
            const __bf16* ap = &hc[mrow * H_ + ke * 32 + k_half];
            v16bf a;
            ((bf16x8*)&a)[0] = *(const bf16x8*)(ap);
            ((bf16x8*)&a)[1] = *(const bf16x8*)(ap + 8);

            #pragma unroll
            for (int j = 0; j < 4; ++j) {
                acc[j] = __builtin_amdgcn_wmma_f32_16x16x32_bf16(
                    false, a, false, bbuf2[cur][j], (short)0, acc[j],
                    false, false);
            }
        }

        // fused epilogue: x_proj(K=2) + b_in + b_hh + sigma*noise, ReLU
        const float* xin = xin_s[t & 1];
        #pragma unroll
        for (int j = 0; j < 4; ++j) {
            const int n = wv * 64 + j * 16 + (lane & 15);
            const float w0 = win_s[n * 2 + 0];
            const float w1 = win_s[n * 2 + 1];
            const float bb = bin_s[n] + bhh_s[n];
            #pragma unroll
            for (int r = 0; r < 8; ++r) {
                const int m = r + mb;
                const size_t gi = ((size_t)(b0 + m) * T_ + t) * H_ + n;
                float v = acc[j][r]
                        + xin[m * 2 + 0] * w0 + xin[m * 2 + 1] * w1
                        + bb + sig * nz[j][r];
                v = fmaxf(v, 0.0f);
                net[gi] = v;                       // net_units (f32, to HBM)
                hn[m * H_ + n] = (__bf16)v;        // h(t+1) (bf16, LDS)
            }
        }
        // One barrier/step: h(t+1) writes visible, and all waves are done
        // reading h(t) before it becomes next step's write buffer.
        __syncthreads();
    }
}

// ---------------------------------------------------------------------------
// read_out_units = net @ W_fc^T + b_fc  (O=2), plus out = last-step readout
// ---------------------------------------------------------------------------
__global__ void readout_kernel(const float* __restrict__ net,
                               const float* __restrict__ Wfc,  // [2,H]
                               const float* __restrict__ bfc,  // [2]
                               float* __restrict__ ro,         // [B,T,2]
                               float* __restrict__ out)        // [B,2]
{
    int bt = blockIdx.x * blockDim.x + threadIdx.x;
    if (bt >= B_ * T_) return;
    const float4* n4 = (const float4*)(net + (size_t)bt * H_);
    const float4* w0 = (const float4*)(Wfc);
    const float4* w1 = (const float4*)(Wfc + H_);
    float a0 = 0.f, a1 = 0.f;
    #pragma unroll 4
    for (int k = 0; k < H_ / 4; ++k) {
        float4 v = n4[k], x = w0[k], y = w1[k];
        a0 += v.x * x.x + v.y * x.y + v.z * x.z + v.w * x.w;
        a1 += v.x * y.x + v.y * y.y + v.z * y.z + v.w * y.w;
    }
    a0 += bfc[0]; a1 += bfc[1];
    ro[bt * 2 + 0] = a0;
    ro[bt * 2 + 1] = a1;
    if ((bt % T_) == (T_ - 1)) {
        int b = bt / T_;
        out[b * 2 + 0] = a0;
        out[b * 2 + 1] = a1;
    }
}

__global__ void hidden_copy_kernel(const float* __restrict__ net,
                                   float* __restrict__ hidden) {
    int i = blockIdx.x * blockDim.x + threadIdx.x;  // B*H
    if (i >= B_ * H_) return;
    int b = i / H_, h = i % H_;
    hidden[i] = net[((size_t)b * T_ + (T_ - 1)) * H_ + h];
}

// ---------------------------------------------------------------------------
extern "C" void kernel_launch(void* const* d_in, const int* in_sizes, int n_in,
                              void* d_out, int out_size, void* d_ws, size_t ws_size,
                              hipStream_t stream) {
    const float* input = (const float*)d_in[0];   // [B,T,2]
    const float* sigma = (const float*)d_in[1];   // [1]
    const float* noise = (const float*)d_in[2];   // [B,T,H]
    const float* W_in  = (const float*)d_in[3];   // [H,2]
    const float* b_in  = (const float*)d_in[4];   // [H]
    const float* W_hh  = (const float*)d_in[5];   // [H,H]
    const float* b_hh  = (const float*)d_in[6];   // [H]
    const float* W_fc  = (const float*)d_in[7];   // [O,H]
    const float* b_fc  = (const float*)d_in[8];   // [O]

    float* out    = (float*)d_out + OFF_OUT;
    float* hidden = (float*)d_out + OFF_HID;
    float* net    = (float*)d_out + OFF_NET;
    float* ro     = (float*)d_out + OFF_RO;

    __bf16* Wp = (__bf16*)d_ws;                   // 512 KB packed W_hh

    pack_whh_kernel<<<(32 * 16 * 32 * 16) / 256, 256, 0, stream>>>(W_hh, Wp);

    rnn_scan_kernel<<<B_ / 16, 256, 0, stream>>>(
        input, sigma, noise, W_in, b_in, b_hh, Wp, net);

    readout_kernel<<<(B_ * T_ + 255) / 256, 256, 0, stream>>>(
        net, W_fc, b_fc, ro, out);

    hidden_copy_kernel<<<(B_ * H_ + 255) / 256, 256, 0, stream>>>(net, hidden);
}
